// BDHAttention_28587302322334
// MI455X (gfx1250) — compile-verified, matrix operand order checked by reading, not run on hardware
//
#include <hip/hip_runtime.h>
#include <hip/hip_bf16.h>

// ---------------------------------------------------------------------------
// Sizes fixed by the reference: B=2, H=12, T=2048, N=256
// ---------------------------------------------------------------------------
#define BB   2
#define HH   12
#define BH   (BB*HH)
#define TT   2048
#define NN   256
#define TPW  16          // t-rows per wave
#define BM   128         // t-rows per workgroup (8 waves)

typedef __attribute__((ext_vector_type(16))) _Float16 v16h;
typedef __attribute__((ext_vector_type(8)))  _Float16 v8h;
typedef __attribute__((ext_vector_type(8)))  float    v8f;

// ---------------------------------------------------------------------------
// Kernel 1: RoPE on Q and K, fp32 -> fp16, into workspace.
// theta = 2^16 => freq(n) = 2^(-(n&~1)/16) / (2*pi); phase = t*freq;
// ph = frac(phase)*2*pi; out_even = v_e*c - v_o*s; out_odd = v_o*c + v_e*s.
// One thread per (bh*T row, pair). 128 pairs per row.
// ---------------------------------------------------------------------------
__global__ __launch_bounds__(256) void rope_f16_kernel(
    const float* __restrict__ Q, const float* __restrict__ K,
    _Float16* __restrict__ QR, _Float16* __restrict__ KR)
{
    size_t idx = (size_t)blockIdx.x * blockDim.x + threadIdx.x; // BH*T*128 total
    int    p   = (int)(idx & 127);          // pair index, n = 2p
    size_t row = idx >> 7;                  // bh*T + t
    int    t   = (int)(row & (TT - 1));

    float q     = (float)(2 * p);                       // floor(n/2)*2
    float freq  = exp2f(-q * (1.0f / 16.0f)) * 0.15915494309189535f; // /2pi
    float phase = (float)t * freq;
    phase      -= floorf(phase);
    float ang   = phase * 6.283185307179586f;
    float s     = __sinf(ang);
    float c     = __cosf(ang);

    size_t base = row * NN + (size_t)(2 * p);
    float q0 = Q[base], q1 = Q[base + 1];
    float k0 = K[base], k1 = K[base + 1];
    QR[base]     = (_Float16)(q0 * c - q1 * s);
    QR[base + 1] = (_Float16)(q1 * c + q0 * s);
    KR[base]     = (_Float16)(k0 * c - k1 * s);
    KR[base + 1] = (_Float16)(k1 * c + k0 * s);
}

// ---------------------------------------------------------------------------
// Kernel 2: V [bh][T][N] fp32 -> Vt [bh][N][T] fp16 (LDS-tiled transpose)
// 64x64 tiles, 256 threads.
// ---------------------------------------------------------------------------
__global__ __launch_bounds__(256) void vtrans_f16_kernel(
    const float* __restrict__ V, _Float16* __restrict__ VT)
{
    __shared__ _Float16 tile[64][72];   // padded to dodge bank conflicts
    const int bh = blockIdx.z;
    const int t0 = blockIdx.x * 64;
    const int n0 = blockIdx.y * 64;

    const float* src = V + ((size_t)bh * TT + t0) * NN + n0;
    int ln = threadIdx.x & 63;   // n within tile (coalesced load)
    int lr = threadIdx.x >> 6;   // 0..3
    #pragma unroll
    for (int j = 0; j < 16; ++j) {
        int tl = lr + 4 * j;
        tile[tl][ln] = (_Float16)src[(size_t)tl * NN + ln];
    }
    __syncthreads();

    _Float16* dst = VT + ((size_t)bh * NN + n0) * TT + t0;
    int wt = threadIdx.x & 63;   // t within tile (coalesced store)
    int wr = threadIdx.x >> 6;
    #pragma unroll
    for (int j = 0; j < 16; ++j) {
        int nl = wr + 4 * j;
        dst[(size_t)nl * TT + wt] = tile[wt][nl];
    }
}

// ---------------------------------------------------------------------------
// Fragment loader: row-major 16x32 f16 tile in WMMA A-operand layout.
// lane m (0-15) holds row m, K chunks {kb..kb+7, kb+16..kb+23}, kb = 8*(lane>=16)
// (row for lanes 16-31 is lane-16). Two global_load_b128 per lane.
// ---------------------------------------------------------------------------
__device__ __forceinline__ v16h load_frag(const _Float16* rowbase, int ld,
                                          int m, int kb)
{
    const _Float16* p = rowbase + (size_t)m * ld + kb;
    v8h lo = *(const v8h*)(p);
    v8h hi = *(const v8h*)(p + 16);
    v16h r;
    #pragma unroll
    for (int i = 0; i < 8; ++i) { r[i] = lo[i]; r[i + 8] = hi[i]; }
    return r;
}

__device__ __forceinline__ v8f wmma_f16(v16h a, v16h b, v8f c)
{
    return __builtin_amdgcn_wmma_f32_16x16x32_f16(
        /*neg_a=*/false, a, /*neg_b=*/false, b,
        /*c_mod=*/(short)0, c, /*reuse_a=*/false, /*reuse_b=*/false);
}

// ---------------------------------------------------------------------------
// Kernel 3: strictly-causal linear attention.
//   O[t,:] = sum_{s<t} (QR[t]·KR[s]) * V[s,:]
// Grid (T/128, BH); 8 waves/block; wave owns 16 t-rows.
// First GEMM computed transposed: St[s,t] = KR·QRt so that its f32 C-layout
// converts in-lane into the f16 A-operand of the second GEMM O += S·V.
// ---------------------------------------------------------------------------
__global__ __launch_bounds__(256) void causal_linattn_kernel(
    const _Float16* __restrict__ QR, const _Float16* __restrict__ KR,
    const _Float16* __restrict__ VT, float* __restrict__ out)
{
    const int bh   = blockIdx.y;
    const int wave = threadIdx.x >> 5;
    const int lane = threadIdx.x & 31;
    const int t0   = blockIdx.x * BM + wave * TPW;

    const size_t off = (size_t)bh * TT * NN;
    const _Float16* qbase = QR + off + (size_t)t0 * NN;
    const _Float16* kbase = KR + off;
    const _Float16* vbase = VT + off;      // [N][T]

    const int m  = lane & 15;
    const int kb = (lane >> 4) << 3;       // 0 or 8

    // Hoist QR fragments (B operand of the St GEMM): 8 n-chunks of 32.
    v16h qf[8];
    #pragma unroll
    for (int c = 0; c < 8; ++c)
        qf[c] = load_frag(qbase + c * 32, NN, m, kb);

    // Output accumulators: 16 n-tiles of 16 -> full 16x256 row block in f32.
    v8f o[16];
    #pragma unroll
    for (int i = 0; i < 16; ++i)
        #pragma unroll
        for (int r = 0; r < 8; ++r) o[i][r] = 0.0f;

    // Stream s in chunks of 32 (strict causal: need s <= t0+14).
    for (int sc = 0; sc < t0 + 15; sc += 32) {
        // St tiles: St[s,t], s-subtiles [sc,sc+16) and [sc+16,sc+32).
        v8f st0, st1;
        #pragma unroll
        for (int r = 0; r < 8; ++r) { st0[r] = 0.0f; st1[r] = 0.0f; }

        #pragma unroll
        for (int c = 0; c < 8; ++c) {
            v16h kf0 = load_frag(kbase + (size_t)sc * NN + c * 32, NN, m, kb);
            st0 = wmma_f16(kf0, qf[c], st0);
            v16h kf1 = load_frag(kbase + (size_t)(sc + 16) * NN + c * 32, NN, m, kb);
            st1 = wmma_f16(kf1, qf[c], st1);
        }

        // Strict-causal mask on diagonal-overlapping chunks (wave-uniform).
        // St element (vgpr r, lane): s = base + r, t = t0 + m; keep iff s < t.
        if (sc + 31 >= t0) {
            const int t   = t0 + m;
            const int sb0 = sc + kb;
            const int sb1 = sc + 16 + kb;
            #pragma unroll
            for (int r = 0; r < 8; ++r) {
                if (sb0 + r >= t) st0[r] = 0.0f;
                if (sb1 + r >= t) st1[r] = 0.0f;
            }
        }

        // In-lane repack: St f32 C-layout -> f16 A-operand of S (16t x 32s).
        v16h sa;
        #pragma unroll
        for (int r = 0; r < 8; ++r) {
            sa[r]     = (_Float16)st0[r];
            sa[r + 8] = (_Float16)st1[r];
        }

        // O[t, n-tile] += S * V-chunk ; B columns are rows of Vt (contiguous).
        #pragma unroll
        for (int i = 0; i < 16; ++i) {
            v16h vf = load_frag(vbase + (size_t)(i * 16) * TT + sc, TT, m, kb);
            o[i] = wmma_f16(sa, vf, o[i]);
        }
    }

    // Store: D element (tile i, vgpr r, lane): t = t0 + kb + r, n = 16*i + m.
    const int trow = t0 + kb;
    float* obase = out + off;
    #pragma unroll
    for (int i = 0; i < 16; ++i)
        #pragma unroll
        for (int r = 0; r < 8; ++r)
            obase[(size_t)(trow + r) * NN + i * 16 + m] = o[i][r];
}

// ---------------------------------------------------------------------------
// Launch
// ---------------------------------------------------------------------------
extern "C" void kernel_launch(void* const* d_in, const int* in_sizes, int n_in,
                              void* d_out, int out_size, void* d_ws, size_t ws_size,
                              hipStream_t stream)
{
    const float* Q = (const float*)d_in[0];
    const float* K = (const float*)d_in[1];
    const float* V = (const float*)d_in[2];
    float* out = (float*)d_out;

    // Workspace: QR, KR, Vt in f16 (3 * 24*2048*256 * 2B ~= 75.5 MB)
    const size_t per = (size_t)BH * TT * NN;
    _Float16* QRw = (_Float16*)d_ws;
    _Float16* KRw = QRw + per;
    _Float16* VTw = KRw + per;

    // 1) RoPE(Q), RoPE(K) -> f16
    {
        size_t total = (size_t)BH * TT * (NN / 2);        // one thread per pair
        dim3 grid((unsigned)(total / 256));
        rope_f16_kernel<<<grid, 256, 0, stream>>>(Q, K, QRw, KRw);
    }
    // 2) V -> Vt f16
    {
        dim3 grid(TT / 64, NN / 64, BH);
        vtrans_f16_kernel<<<grid, 256, 0, stream>>>(V, VTw);
    }
    // 3) causal linear attention (WMMA)
    {
        dim3 grid(TT / BM, BH);
        causal_linattn_kernel<<<grid, 256, 0, stream>>>(QRw, KRw, VTw, out);
    }
}
